// MHSA_90872918049357
// MI455X (gfx1250) — matrix operand order repaired and checked by reading, not compiled, not run on hardware
//
#include <hip/hip_runtime.h>
#include <hip/hip_bf16.h>

typedef __attribute__((ext_vector_type(16))) _Float16 v16h;
typedef __attribute__((ext_vector_type(8)))  _Float16 h8;
typedef __attribute__((ext_vector_type(8)))  float    v8f;

union V16H { v16h v; h8 h[2]; };

// ---- WMMA helpers -----------------------------------------------------------
__device__ inline v8f wmma_f16(v16h a, v16h b, v8f c) {
    // D = A(16x32 f16) * B(32x16 f16) + C(16x16 f32)
    return __builtin_amdgcn_wmma_f32_16x16x32_f16(
        false, a, false, b, (short)0, c, false, false);
}

// A-fragment: lane (l16 = lane&15) holds row M=l16.
// lanes 0-15: halves 0..15 = K kbase+0..15
// lanes16-31: halves 0..15 = K kbase+16..31
__device__ inline v16h load_a_frag(const _Float16* __restrict__ row, int kbase, int dlo) {
    V16H u;
    u.h[0] = *(const h8*)(row + kbase + dlo * 2);
    u.h[1] = *(const h8*)(row + kbase + dlo * 2 + 8);
    return u.v;
}

// B-fragment: lane holds column N=lane&15; 16 contiguous K values
// starting at kbase (caller folds in (lane>>4)*16).
__device__ inline v16h load_b_frag(const _Float16* __restrict__ row, int kbase) {
    V16H u;
    u.h[0] = *(const h8*)(row + kbase);
    u.h[1] = *(const h8*)(row + kbase + 8);
    return u.v;
}

// async copy 16B global -> LDS, tracked by ASYNCcnt
__device__ inline void async_b128(unsigned lds_off, const void* gaddr) {
    asm volatile("global_load_async_to_lds_b128 %0, %1, off"
                 :: "v"(lds_off), "v"((unsigned long long)gaddr)
                 : "memory");
}
__device__ inline void wait_async0() {
    asm volatile("s_wait_asynccnt 0x0" ::: "memory");
}

// ---- small conversion / repack kernels -------------------------------------
__global__ void f32_to_f16_kernel(const float* __restrict__ in,
                                  _Float16* __restrict__ out, int n) {
    int i = blockIdx.x * blockDim.x + threadIdx.x;
    if (i < n) out[i] = (_Float16)in[i];
}

// x [B,C,L] f32 -> xh [B,L,C] f16 (rows over l for the projection GEMM)
__global__ void transpose_x_kernel(const float* __restrict__ x,
                                   _Float16* __restrict__ xh,
                                   int B, int C, int L) {
    int i = blockIdx.x * blockDim.x + threadIdx.x;
    if (i >= B * C * L) return;
    int c = i % C;
    int l = (i / C) % L;
    int b = i / (C * L);
    xh[((size_t)b * L + l) * C + c] = (_Float16)x[((size_t)b * C + c) * L + l];
}

// P f32 [B*L, nH*D] (row l, col h*D+d) -> l2-normalized f16 [B,nH,L,D]
__global__ void norm_repack_kernel(const float* __restrict__ P,
                                   _Float16* __restrict__ Ph,
                                   int B, int nH, int L, int D) {
    int t = blockIdx.x * blockDim.x + threadIdx.x;
    if (t >= B * nH * L) return;
    int l = t % L;
    int h = (t / L) % nH;
    int b = t / (L * nH);
    const float* src = P + ((size_t)b * L + l) * (nH * D) + h * D;
    float s = 0.f;
    for (int d = 0; d < D; ++d) { float v = src[d]; s += v * v; }
    float inv = 1.0f / fmaxf(sqrtf(s), 1e-6f);
    _Float16* dst = Ph + ((size_t)(b * nH + h) * L + l) * D;
    for (int d = 0; d < D; ++d) dst[d] = (_Float16)(src[d] * inv);
}

// V f32 [B*L, nH*D] -> f16 [B,nH,D,L]  (j contiguous for B-fragment loads)
__global__ void v_repack_kernel(const float* __restrict__ P,
                                _Float16* __restrict__ Ph,
                                int B, int nH, int L, int D) {
    int t = blockIdx.x * blockDim.x + threadIdx.x;
    if (t >= B * nH * D * L) return;
    int l = t % L;
    int v = (t / L) % D;
    int h = (t / (L * D)) % nH;
    int b = t / (L * D * nH);
    Ph[((size_t)(b * nH + h) * D + v) * L + l] =
        (_Float16)P[((size_t)b * L + l) * (nH * D) + h * D + v];
}

// ---- generic WMMA GEMM: out[M,N] = A[M,K] * W[N,K]^T + bias (+residual) ----
// mode 0: out row-major [M,N].  mode 1: out[(b*N + col)*Lsp + l], row = b*Lsp+l
__global__ void __launch_bounds__(128) gemm_wmma_kernel(
    const _Float16* __restrict__ A, const _Float16* __restrict__ W,
    const float* __restrict__ bias, float* __restrict__ out,
    const float* __restrict__ residual,
    int M, int N, int K, int mode, int Lsp) {
    const int lane = threadIdx.x & 31;
    const int wave = threadIdx.x >> 5;
    const int tile = blockIdx.x * 4 + wave;
    const int tiles_n = N >> 4;
    const int m0 = (tile / tiles_n) << 4;
    const int n0 = (tile % tiles_n) << 4;
    if (m0 >= M) return;  // wave-uniform

    const int half = lane >> 4;
    const int l16  = lane & 15;
    const int dlo  = half * 8;

    const _Float16* arow = A + (size_t)(m0 + l16) * K;
    const _Float16* wrow = W + (size_t)(n0 + l16) * K;

    v8f c = {0.f, 0.f, 0.f, 0.f, 0.f, 0.f, 0.f, 0.f};
    for (int k = 0; k < K; k += 32) {
        v16h a = load_a_frag(arow, k, dlo);
        v16h b = load_b_frag(wrow, k + half * 16);
        c = wmma_f16(a, b, c);
    }

#pragma unroll
    for (int r = 0; r < 8; ++r) {
        int row = m0 + r + 8 * half;
        int col = n0 + l16;
        float v = c[r] + bias[col];
        size_t addr;
        if (mode == 0) {
            addr = (size_t)row * N + col;
        } else {
            int bb = row / Lsp;
            int l  = row % Lsp;
            addr = ((size_t)bb * N + col) * Lsp + l;
        }
        if (residual) v += residual[addr];
        out[addr] = v;
    }
}

// ---- flash attention --------------------------------------------------------
// One block = 4 waves = 64 i-rows of one (b,h). K/V streamed in 64-j chunks,
// cooperatively staged into double-buffered LDS with async loads (ASYNCcnt),
// overlapped with WMMA compute. 16 WMMAs per wave per chunk.
__global__ void __launch_bounds__(128) flash_attn_kernel(
    const _Float16* __restrict__ Qh, const _Float16* __restrict__ Kh,
    const _Float16* __restrict__ Vh, _Float16* __restrict__ Rh) {
    constexpr int L  = 4096;
    constexpr int nH = 4;
    constexpr int D  = 64;
    constexpr int JC = 64;          // j-chunk
    constexpr int NCH = L / JC;     // 64 chunks

    __shared__ __align__(16) _Float16 kbuf[2][JC * D];   // [j][d]
    __shared__ __align__(16) _Float16 vbuf[2][D * JC];   // [v][j]
    __shared__ __align__(16) _Float16 pbuf[4][16 * JC];  // per-wave P staging

    const int tid  = threadIdx.x;
    const int lane = tid & 31;
    const int wave = tid >> 5;
    const int bid  = blockIdx.x;
    const int iblk = bid & 63;
    const int h    = (bid >> 6) & 3;
    const int b    = bid >> 8;
    const int i0   = iblk * 64 + wave * 16;

    const int half = lane >> 4;
    const int l16  = lane & 15;
    const int dlo  = half * 8;

    const _Float16* Qbase = Qh + (size_t)(b * nH + h) * L * D;
    const _Float16* Kbase = Kh + (size_t)(b * nH + h) * L * D;
    const _Float16* Vbase = Vh + (size_t)(b * nH + h) * D * L;

    // Q A-fragments, resident for the whole j-loop
    const _Float16* qrow = Qbase + (size_t)(i0 + l16) * D;
    const v16h qa0 = load_a_frag(qrow, 0,  dlo);
    const v16h qa1 = load_a_frag(qrow, 32, dlo);

    float m_run[8], l_run[8];
    v8f o[4];
#pragma unroll
    for (int r = 0; r < 8; ++r) { m_run[r] = -1e30f; l_run[r] = 0.f; }
#pragma unroll
    for (int n = 0; n < 4; ++n) o[n] = (v8f){0.f,0.f,0.f,0.f,0.f,0.f,0.f,0.f};

    // cooperative async stage of one 64-j chunk of K and V into LDS buffer bufi
    auto stage = [&](int jc, int bufi) {
        const int j0 = jc * JC;
#pragma unroll
        for (int i = 0; i < 4; ++i) {
            int idx = i * 128 + tid;     // 0..511 16B segments per matrix
            int row = idx >> 3;          // 0..63
            int seg = (idx & 7) * 8;     // half offset within row
            async_b128((unsigned)(size_t)(&kbuf[bufi][row * D + seg]),
                       Kbase + (size_t)(j0 + row) * D + seg);
            async_b128((unsigned)(size_t)(&vbuf[bufi][row * JC + seg]),
                       Vbase + (size_t)row * L + j0 + seg);
        }
    };

    stage(0, 0);
    wait_async0();
    __syncthreads();

    _Float16* pw = &pbuf[wave][0];

    for (int jc = 0; jc < NCH; ++jc) {
        const int cur = jc & 1;
        if (jc + 1 < NCH) stage(jc + 1, cur ^ 1);  // overlap with compute

        // ---- S = Q·K^T: four 16x16 tiles over the 64-j chunk (K-dim = d) ---
        v8f s[4];
#pragma unroll
        for (int t = 0; t < 4; ++t) {
            const _Float16* kr = &kbuf[cur][(t * 16 + l16) * D];
            v8f acc = {0.f,0.f,0.f,0.f,0.f,0.f,0.f,0.f};
            acc = wmma_f16(qa0, load_b_frag(kr, 0  + half * 16), acc);
            acc = wmma_f16(qa1, load_b_frag(kr, 32 + half * 16), acc);
            s[t] = acc;
        }

        // ---- streaming softmax over 64 j's (reductions amortized 4 tiles) --
#pragma unroll
        for (int r = 0; r < 8; ++r) {
            float mx = fmaxf(fmaxf(s[0][r], s[1][r]), fmaxf(s[2][r], s[3][r]));
#pragma unroll
            for (int mk = 1; mk <= 8; mk <<= 1)
                mx = fmaxf(mx, __shfl_xor(mx, mk, 32));
            float mnew  = fmaxf(m_run[r], mx);
            float scale = __expf(m_run[r] - mnew);
            float rs = 0.f;
            float e[4];
#pragma unroll
            for (int t = 0; t < 4; ++t) { e[t] = __expf(s[t][r] - mnew); rs += e[t]; }
#pragma unroll
            for (int mk = 1; mk <= 8; mk <<= 1)
                rs += __shfl_xor(rs, mk, 32);
            l_run[r] = l_run[r] * scale + rs;
            m_run[r] = mnew;
#pragma unroll
            for (int t = 0; t < 4; ++t) s[t][r] = e[t];  // s becomes P
#pragma unroll
            for (int n = 0; n < 4; ++n) o[n][r] *= scale;
        }

        // ---- transpose P (C-layout -> A-layout) via wave-private LDS -------
#pragma unroll
        for (int r = 0; r < 8; ++r) {
            int row = r + 8 * half;
#pragma unroll
            for (int t = 0; t < 4; ++t)
                pw[row * JC + t * 16 + l16] = (_Float16)s[t][r];
        }

        // ---- O += P·V: two 32-j k-steps x four 16-v N-tiles ---------------
#pragma unroll
        for (int kk = 0; kk < 2; ++kk) {
            V16H ua;  // DS ops same-wave are in-order: safe read-after-write
            ua.h[0] = *(const h8*)(pw + l16 * JC + kk * 32 + dlo * 2);
            ua.h[1] = *(const h8*)(pw + l16 * JC + kk * 32 + dlo * 2 + 8);
            v16h pa = ua.v;
#pragma unroll
            for (int n = 0; n < 4; ++n) {
                const _Float16* vr = &vbuf[cur][(n * 16 + l16) * JC];
                v16h vb = load_b_frag(vr, kk * 32 + half * 16);
                o[n] = wmma_f16(pa, vb, o[n]);
            }
        }

        // my next-chunk loads done; barrier => everyone's done + readers done
        wait_async0();
        __syncthreads();
    }

    // ---- epilogue: normalize by l_run, write Rh [B*L, nH*dv] f16 -----------
#pragma unroll
    for (int n = 0; n < 4; ++n) {
#pragma unroll
        for (int r = 0; r < 8; ++r) {
            int row = i0 + r + 8 * half;
            int col = h * 64 + n * 16 + l16;
            Rh[((size_t)b * L + row) * 256 + col] =
                (_Float16)(o[n][r] / l_run[r]);
        }
    }
}

// ---- host-side orchestration ------------------------------------------------
extern "C" void kernel_launch(void* const* d_in, const int* in_sizes, int n_in,
                              void* d_out, int out_size, void* d_ws, size_t ws_size,
                              hipStream_t stream) {
    constexpr int B = 2, C = 256, Hh = 64, Ww = 64, nH = 4, DK = 64, DV = 64;
    constexpr int L = Hh * Ww;   // 4096
    constexpr int M = B * L;     // 8192 GEMM rows

    const float* x  = (const float*)d_in[0];
    const float* Wq = (const float*)d_in[1];
    const float* bq = (const float*)d_in[2];
    const float* Wk = (const float*)d_in[3];
    const float* bk = (const float*)d_in[4];
    const float* Wv = (const float*)d_in[5];
    const float* bv = (const float*)d_in[6];
    const float* Wm = (const float*)d_in[7];
    const float* bm = (const float*)d_in[8];
    float* out = (float*)d_out;

    char* ws = (char*)d_ws;
    size_t off = 0;
    auto alloc = [&](size_t bytes) -> void* {
        void* p = ws + off;
        off += (bytes + 255) & ~(size_t)255;
        return p;
    };
    _Float16* xh   = (_Float16*)alloc((size_t)M * C * 2);
    _Float16* wq16 = (_Float16*)alloc((size_t)C * C * 2);
    _Float16* wk16 = (_Float16*)alloc((size_t)C * C * 2);
    _Float16* wv16 = (_Float16*)alloc((size_t)C * C * 2);
    _Float16* wm16 = (_Float16*)alloc((size_t)C * C * 2);
    float*    Qf   = (float*)   alloc((size_t)M * C * 4);
    float*    Kf   = (float*)   alloc((size_t)M * C * 4);
    float*    Vf   = (float*)   alloc((size_t)M * C * 4);
    _Float16* Qh   = (_Float16*)alloc((size_t)M * C * 2);
    _Float16* Kh   = (_Float16*)alloc((size_t)M * C * 2);
    _Float16* Vh   = (_Float16*)alloc((size_t)M * C * 2);
    _Float16* Rh   = (_Float16*)alloc((size_t)M * C * 2);
    (void)ws_size; (void)in_sizes; (void)n_in; (void)out_size;

    int n;
    // 1. convert x (transposed) and weights to f16
    n = B * C * L;
    transpose_x_kernel<<<(n + 255) / 256, 256, 0, stream>>>(x, xh, B, C, L);
    n = C * C;
    f32_to_f16_kernel<<<(n + 255) / 256, 256, 0, stream>>>(Wq, wq16, n);
    f32_to_f16_kernel<<<(n + 255) / 256, 256, 0, stream>>>(Wk, wk16, n);
    f32_to_f16_kernel<<<(n + 255) / 256, 256, 0, stream>>>(Wv, wv16, n);
    f32_to_f16_kernel<<<(n + 255) / 256, 256, 0, stream>>>(Wm, wm16, n);

    // 2. QKV projections: [8192,256] = xh[8192,256] * W[256,256]^T + bias
    const int gemm_blocks = (M / 16) * (C / 16) / 4;  // 4 waves/block, 1 tile/wave
    gemm_wmma_kernel<<<gemm_blocks, 128, 0, stream>>>(xh, wq16, bq, Qf, nullptr, M, C, C, 0, L);
    gemm_wmma_kernel<<<gemm_blocks, 128, 0, stream>>>(xh, wk16, bk, Kf, nullptr, M, C, C, 0, L);
    gemm_wmma_kernel<<<gemm_blocks, 128, 0, stream>>>(xh, wv16, bv, Vf, nullptr, M, C, C, 0, L);

    // 3. l2-normalize Q,K over d; repack V transposed for B-fragments
    n = B * nH * L;
    norm_repack_kernel<<<(n + 255) / 256, 256, 0, stream>>>(Qf, Qh, B, nH, L, DK);
    norm_repack_kernel<<<(n + 255) / 256, 256, 0, stream>>>(Kf, Kh, B, nH, L, DK);
    n = B * nH * DV * L;
    v_repack_kernel<<<(n + 255) / 256, 256, 0, stream>>>(Vf, Vh, B, nH, L, DV);

    // 4. flash attention: 512 blocks x 4 waves, LDS-staged K/V, async pipelined
    flash_attn_kernel<<<B * nH * (L / 64), 128, 0, stream>>>(Qh, Kh, Vh, Rh);

    // 5. output projection + bias + residual, written [B,C,L]
    gemm_wmma_kernel<<<gemm_blocks, 128, 0, stream>>>(Rh, wm16, bm, out, x, M, C, C, 1, L);
}
